// GraphAttentionLayer_4784593567901
// MI455X (gfx1250) — compile-verified, matrix operand order checked by reading, not run on hardware
//
#include <hip/hip_runtime.h>
#include <hip/hip_bf16.h>

#define BATCH 4
#define NN    4096
#define FIN   256
#define FOUT  64
#define ALPHA 0.2f
#define NEG_BIG -9.0e15f
#define CHUNK 128               // adj columns staged per TDM tile

typedef __attribute__((ext_vector_type(16))) _Float16     v16h;
typedef __attribute__((ext_vector_type(8)))  _Float16     v8h;
typedef __attribute__((ext_vector_type(8)))  float        v8f;
typedef __attribute__((ext_vector_type(4)))  float        f32x4;
typedef __attribute__((ext_vector_type(4)))  unsigned int u32x4;
typedef __attribute__((ext_vector_type(8)))  unsigned int u32x8;

// ---------------------------------------------------------------------------
// Kernel 0: W [256][64] f32 -> WT [64][256] f16 (so B-fragments load contiguous)
// ---------------------------------------------------------------------------
__global__ void wt_transpose_kernel(const float* __restrict__ W,
                                    _Float16* __restrict__ WT) {
  int idx = blockIdx.x * blockDim.x + threadIdx.x;  // 64*256 = 16384
  int n = idx >> 8;          // 0..63  (Fout)
  int k = idx & 255;         // 0..255 (Fin)
  WT[(size_t)n * FIN + k] = (_Float16)W[(size_t)k * FOUT + n];
}

// ---------------------------------------------------------------------------
// Kernel 1: h = feat @ W  via v_wmma_f32_16x16x32_f16.
// One wave computes one [16 x 64] tile of h (4 N-tiles, K-loop of 8).
// feat is read once (16 MB) -> non-temporal loads so it never pollutes L2.
// ---------------------------------------------------------------------------
__global__ __launch_bounds__(128) void proj_kernel(const float* __restrict__ feat,
                                                   const _Float16* __restrict__ WT,
                                                   float* __restrict__ h) {
  int wave = threadIdx.x >> 5;
  int lane = threadIdx.x & 31;
  int half = lane >> 4;
  int mrow = lane & 15;
  int g0   = half * 8;

  int tile = blockIdx.x * 4 + wave;      // 0..1023 = BATCH * (NN/16)
  int b    = tile >> 8;
  int row0 = (tile & 255) * 16;

  v8f acc[4];
#pragma unroll
  for (int t = 0; t < 4; ++t)
#pragma unroll
    for (int r = 0; r < 8; ++r) acc[t][r] = 0.0f;

  const float* arow = feat + ((size_t)(b * NN + row0 + mrow)) * FIN;

  for (int kc = 0; kc < FIN; kc += 32) {
    const f32x4* f0p = reinterpret_cast<const f32x4*>(arow + kc + g0);
    f32x4 f0 = __builtin_nontemporal_load(f0p);
    f32x4 f1 = __builtin_nontemporal_load(f0p + 1);
    const f32x4* f2p = reinterpret_cast<const f32x4*>(arow + kc + 16 + g0);
    f32x4 f2 = __builtin_nontemporal_load(f2p);
    f32x4 f3 = __builtin_nontemporal_load(f2p + 1);

    v16h a;
#pragma unroll
    for (int i = 0; i < 4; ++i) {
      a[i]      = (_Float16)f0[i];
      a[4 + i]  = (_Float16)f1[i];
      a[8 + i]  = (_Float16)f2[i];
      a[12 + i] = (_Float16)f3[i];
    }

#pragma unroll
    for (int t = 0; t < 4; ++t) {
      const _Float16* wrow = WT + (size_t)(16 * t + mrow) * FIN + kc + g0;
      v8h w0 = *reinterpret_cast<const v8h*>(wrow);
      v8h w1 = *reinterpret_cast<const v8h*>(wrow + 16);
      v16h bf;
#pragma unroll
      for (int i = 0; i < 8; ++i) { bf[i] = w0[i]; bf[8 + i] = w1[i]; }
      acc[t] = __builtin_amdgcn_wmma_f32_16x16x32_f16(false, a, false, bf,
                                                      (short)0, acc[t], false, false);
    }
  }

  // D-tile: lane holds col n = 16t + mrow, VGPR r holds row r + 8*half.
#pragma unroll
  for (int t = 0; t < 4; ++t)
#pragma unroll
    for (int r = 0; r < 8; ++r)
      h[((size_t)(b * NN) + row0 + r + 8 * half) * FOUT + 16 * t + mrow] = acc[t][r];
}

// ---------------------------------------------------------------------------
// Kernel 2: l1 = h.a1, l2 = h.a2  (per row), and hT f16 [B][64][4096]
// ---------------------------------------------------------------------------
__global__ void logits_kernel(const float* __restrict__ h,
                              const float* __restrict__ a1,
                              const float* __restrict__ a2,
                              float* __restrict__ l1, float* __restrict__ l2,
                              _Float16* __restrict__ hT) {
  int idx = blockIdx.x * blockDim.x + threadIdx.x;   // BATCH*NN = 16384
  int b = idx >> 12;
  int n = idx & (NN - 1);
  const float* hr = h + (size_t)idx * FOUT;
  float s1 = 0.0f, s2 = 0.0f;
#pragma unroll
  for (int f = 0; f < FOUT; ++f) {
    float v = hr[f];
    s1 += v * a1[f];
    s2 += v * a2[f];
    hT[((size_t)b * FOUT + f) * NN + n] = (_Float16)v;
  }
  l1[idx] = s1;
  l2[idx] = s2;
}

// ---------------------------------------------------------------------------
// Kernel 3: fused flash-style masked attention + P@H (WMMA f16) + ELU.
// The dominant stream (adj, 268 MB) is DMA'd by the Tensor Data Mover into
// double-buffered LDS tiles (16 rows x 128 ints, row stride 4096), tagged NT
// so it never evicts the L2-resident hT. Waves compute from LDS under
// TENSORcnt + workgroup-barrier synchronization.
// ---------------------------------------------------------------------------
__global__ __launch_bounds__(128) void attn_kernel(const int* __restrict__ adj,
                                                   const _Float16* __restrict__ hT,
                                                   const float* __restrict__ l1g,
                                                   const float* __restrict__ l2g,
                                                   float* __restrict__ out) {
  __shared__ int   s_adj[2][16][CHUNK];   // 2 x 8 KB TDM staging buffers
  __shared__ float s_acc[4][16][FOUT];    // 16 KB combine buffer
  __shared__ float s_m[4][16];
  __shared__ float s_l[4][16];

  int wave = threadIdx.x >> 5;
  int lane = threadIdx.x & 31;
  int half = lane >> 4;
  int mrow = lane & 15;
  int g0   = half * 8;

  int b    = blockIdx.x >> 8;
  int row0 = (blockIdx.x & 255) * 16;

  float l1v = l1g[b * NN + row0 + mrow];
  const float* l2row = l2g + (size_t)b * NN;

  // Tensor DMA descriptor pieces (uniform -> SGPRs).
  const unsigned lds_adj0 = (unsigned)(uintptr_t)(&s_adj[0][0][0]);
  const unsigned lds_adj1 = (unsigned)(uintptr_t)(&s_adj[1][0][0]);
  const unsigned long long gbase =
      (unsigned long long)(uintptr_t)(adj + ((size_t)(b * NN) + row0) * NN);

  auto tdm_load_adj = [&](int c) {
    unsigned long long ga = gbase + (unsigned long long)c * (CHUNK * 4);
    u32x4 g0d;
    g0d[0] = 1u;                                           // count=1
    g0d[1] = (c & 1) ? lds_adj1 : lds_adj0;                // lds_addr (bytes)
    g0d[2] = (unsigned)ga;                                 // global_addr[31:0]
    g0d[3] = ((unsigned)(ga >> 32) & 0x01FFFFFFu)          // global_addr[56:32]
             | (2u << 30);                                 // type=2 (image)
    u32x8 g1d;
    g1d[0] = 2u << 16;                                     // data_size=4B
    g1d[1] = ((unsigned)NN & 0xFFFFu) << 16;               // tensor_dim0 lo16
    g1d[2] = ((unsigned)NN >> 16) | (16u << 16);           // dim0 hi | tensor_dim1=16
    g1d[3] = ((unsigned)CHUNK) << 16;                      // tile_dim0=128
    g1d[4] = 16u;                                          // tile_dim1=16, tile_dim2=0
    g1d[5] = (unsigned)NN;                                 // tensor_dim0_stride=4096
    g1d[6] = 0u;
    g1d[7] = 0u;
    asm volatile("tensor_load_to_lds %0, %1 th:TH_LOAD_NT"
                 :: "s"(g0d), "s"(g1d) : "memory");
  };

  float m_run = -3.0e38f;
  float l_run = 0.0f;
  v8f acc[4];
#pragma unroll
  for (int t = 0; t < 4; ++t)
#pragma unroll
    for (int r = 0; r < 8; ++r) acc[t][r] = 0.0f;

  if (wave == 0) tdm_load_adj(0);

  for (int c = 0; c < NN / CHUNK; ++c) {
    if (wave == 0) {
      if (c + 1 < NN / CHUNK) {
        tdm_load_adj(c + 1);                       // prefetch next tile
        __builtin_amdgcn_s_wait_tensorcnt(1);      // in-order: tile c done
      } else {
        __builtin_amdgcn_s_wait_tensorcnt(0);
      }
    }
    __syncthreads();                               // tile c visible to all waves

    int j0 = c * CHUNK + wave * 32;
    const int* abuf = &s_adj[c & 1][mrow][wave * 32 + g0];

    // 16 adjacency ints from LDS + 16 l2 floats, in A-fragment K order
    int4   A0 = *reinterpret_cast<const int4*>(abuf);
    int4   A1 = *reinterpret_cast<const int4*>(abuf + 4);
    int4   A2 = *reinterpret_cast<const int4*>(abuf + 16);
    int4   A3 = *reinterpret_cast<const int4*>(abuf + 20);
    float4 L0 = *reinterpret_cast<const float4*>(l2row + j0 + g0);
    float4 L1 = *reinterpret_cast<const float4*>(l2row + j0 + g0 + 4);
    float4 L2 = *reinterpret_cast<const float4*>(l2row + j0 + 16 + g0);
    float4 L3 = *reinterpret_cast<const float4*>(l2row + j0 + 16 + g0 + 4);

    int   ai[16] = {A0.x, A0.y, A0.z, A0.w, A1.x, A1.y, A1.z, A1.w,
                    A2.x, A2.y, A2.z, A2.w, A3.x, A3.y, A3.z, A3.w};
    float li[16] = {L0.x, L0.y, L0.z, L0.w, L1.x, L1.y, L1.z, L1.w,
                    L2.x, L2.y, L2.z, L2.w, L3.x, L3.y, L3.z, L3.w};

    float s[16];
#pragma unroll
    for (int i = 0; i < 16; ++i) {
      float t = l1v + li[i];
      float e = t > 0.0f ? t : ALPHA * t;           // LeakyReLU
      s[i] = ai[i] > 0 ? e : NEG_BIG;               // adjacency mask
    }

    float tmax = s[0];
#pragma unroll
    for (int i = 1; i < 16; ++i) tmax = fmaxf(tmax, s[i]);
    tmax = fmaxf(tmax, __shfl_xor(tmax, 16, 32));   // combine row halves

    float mnew  = fmaxf(m_run, tmax);
    float scale = __expf(m_run - mnew);

    float psum = 0.0f;
    v16h pa;
#pragma unroll
    for (int i = 0; i < 16; ++i) {
      float p = __expf(s[i] - mnew);
      psum += p;
      pa[i] = (_Float16)p;
    }
    psum += __shfl_xor(psum, 16, 32);
    l_run = l_run * scale + psum;
    m_run = mnew;

    // rescale accumulator rows (row of VGPR r is r + 8*half)
#pragma unroll
    for (int r = 0; r < 8; ++r) {
      float sc = __shfl(scale, r + 8 * half, 32);
#pragma unroll
      for (int t = 0; t < 4; ++t) acc[t][r] *= sc;
    }

    // acc += P[16x32] @ H[32x64]  (4 N-tiles); hT stays hot in L2
#pragma unroll
    for (int t = 0; t < 4; ++t) {
      const _Float16* hrow = hT + ((size_t)b * FOUT + 16 * t + mrow) * NN + j0 + g0;
      v8h w0 = *reinterpret_cast<const v8h*>(hrow);
      v8h w1 = *reinterpret_cast<const v8h*>(hrow + 16);
      v16h bf;
#pragma unroll
      for (int i = 0; i < 8; ++i) { bf[i] = w0[i]; bf[8 + i] = w1[i]; }
      acc[t] = __builtin_amdgcn_wmma_f32_16x16x32_f16(false, pa, false, bf,
                                                      (short)0, acc[t], false, false);
    }

    __syncthreads();   // all waves done with tile c before TDM reuses its buffer
  }

  // publish per-wave partials
#pragma unroll
  for (int t = 0; t < 4; ++t)
#pragma unroll
    for (int r = 0; r < 8; ++r)
      s_acc[wave][r + 8 * half][16 * t + mrow] = acc[t][r];
  if (lane < 16) {
    s_m[wave][mrow] = m_run;
    s_l[wave][mrow] = l_run;
  }
  __syncthreads();

  // combine 4 waves, normalize, ELU, non-temporal store (output written once)
  for (int k = threadIdx.x; k < 16 * FOUT; k += 128) {
    int m = k >> 6;
    int n = k & (FOUT - 1);
    float M = fmaxf(fmaxf(s_m[0][m], s_m[1][m]), fmaxf(s_m[2][m], s_m[3][m]));
    float num = 0.0f, den = 0.0f;
#pragma unroll
    for (int w = 0; w < 4; ++w) {
      float f = __expf(s_m[w][m] - M);
      num += s_acc[w][m][n] * f;
      den += s_l[w][m] * f;
    }
    float v = num / den;
    v = v > 0.0f ? v : (__expf(v) - 1.0f);          // ELU (alpha=1)
    __builtin_nontemporal_store(v, &out[((size_t)(b * NN) + row0 + m) * FOUT + n]);
  }
}

// ---------------------------------------------------------------------------
extern "C" void kernel_launch(void* const* d_in, const int* in_sizes, int n_in,
                              void* d_out, int out_size, void* d_ws, size_t ws_size,
                              hipStream_t stream) {
  const float* feat = (const float*)d_in[0];   // [4,4096,256] f32
  const int*   adj  = (const int*)d_in[1];     // [4,4096,4096] i32
  const float* W    = (const float*)d_in[2];   // [256,64] f32
  const float* a1   = (const float*)d_in[3];   // [64,1] f32
  const float* a2   = (const float*)d_in[4];   // [64,1] f32
  float* out = (float*)d_out;                  // [4,4096,64] f32

  // workspace layout (all offsets 16B-aligned)
  char* ws = (char*)d_ws;
  _Float16* WT = (_Float16*)(ws);                                   // 32 KB
  float*    h  = (float*)(ws + 32768);                              // 4 MB
  _Float16* hT = (_Float16*)(ws + 32768 + 4194304);                 // 2 MB
  float*    l1 = (float*)(ws + 32768 + 4194304 + 2097152);          // 64 KB
  float*    l2 = (float*)(ws + 32768 + 4194304 + 2097152 + 65536);  // 64 KB

  wt_transpose_kernel<<<64, 256, 0, stream>>>(W, WT);
  proj_kernel<<<256, 128, 0, stream>>>(feat, WT, h);
  logits_kernel<<<64, 256, 0, stream>>>(h, a1, a2, l1, l2, hT);
  attn_kernel<<<BATCH * (NN / 16), 128, 0, stream>>>(adj, hT, l1, l2, out);
}
